// TripletGAT_20710332301336
// MI455X (gfx1250) — matrix operand order repaired and checked by reading, not compiled, not run on hardware
//
#include <hip/hip_runtime.h>

typedef __attribute__((ext_vector_type(16))) __bf16 v16bf;
typedef __attribute__((ext_vector_type(8)))  __bf16 v8bf;
typedef __attribute__((ext_vector_type(8)))  float  v8f;

#define NEG_SLOPE 0.2f
#define NEGINF_ENC 0x007FFFFFu   /* ordered-uint encoding of -inf */

static __device__ __forceinline__ unsigned ord_enc(float f) {
  unsigned u = __float_as_uint(f);
  return (u & 0x80000000u) ? ~u : (u | 0x80000000u);
}
static __device__ __forceinline__ float ord_dec(unsigned e) {
  return (e & 0x80000000u) ? __uint_as_float(e & 0x7fffffffu)
                           : __uint_as_float(~e);
}

// ---------------- utility kernels ----------------

__global__ void fill_u32_k(unsigned* p, unsigned v, long long n) {
  long long i = (long long)blockIdx.x * blockDim.x + threadIdx.x;
  if (i < n) p[i] = v;
}

__global__ void cast_bf16_k(const float* __restrict__ in, __bf16* __restrict__ out, long long n) {
  long long i = (long long)blockIdx.x * blockDim.x + threadIdx.x;
  if (i < n) out[i] = (__bf16)in[i];
}

// W: [K, Ncols] row-major f32  ->  Wt: [Ncols, K] row-major bf16
__global__ void transpose_cast_k(const float* __restrict__ W, __bf16* __restrict__ Wt,
                                 int K, int Ncols) {
  int i = blockIdx.x * blockDim.x + threadIdx.x;
  if (i < K * Ncols) {
    int k = i / Ncols, n = i % Ncols;
    Wt[(size_t)n * K + k] = (__bf16)W[i];
  }
}

// ---------------- WMMA GEMM: C[M,Ncols] = A[M,K](bf16) * Bt[Ncols,K]^T ----------------

static __device__ __forceinline__ v16bf load_a_frag(const __bf16* base, int ld, int lane) {
  // A 16x32 bf16 layout: lanes 0-15 hold row m, K = 0..7 (V0-3) & 16..23 (V4-7)
  //                      lanes 16-31 hold row m, K = 8..15 & 24..31
  int m  = lane & 15;
  int kq = lane >> 4;
  const __bf16* p = base + (size_t)m * ld + kq * 8;
  v8bf lo = *(const v8bf*)p;        // K = kq*8 .. kq*8+7
  v8bf hi = *(const v8bf*)(p + 16); // K = 16+kq*8 ..
  return __builtin_shufflevector(lo, hi, 0,1,2,3,4,5,6,7,8,9,10,11,12,13,14,15);
}

static __device__ __forceinline__ v16bf load_b_frag(const __bf16* base, int ld, int lane) {
  // B 32x16 bf16 layout: lane n (0-15) holds column n, K=0..7 & 16..23; lane n+16 holds K=8..15 & 24..31
  int n  = lane & 15;
  int kh = lane >> 4;
  const __bf16* p = base + (size_t)n * ld + kh * 8;
  v8bf lo = *(const v8bf*)p;
  v8bf hi = *(const v8bf*)(p + 16);
  return __builtin_shufflevector(lo, hi, 0,1,2,3,4,5,6,7,8,9,10,11,12,13,14,15);
}

__global__ void gemm_bf16_wmma_k(const __bf16* __restrict__ A, const __bf16* __restrict__ Bt,
                                 float* __restrict__ C, int M, int Ncols, int K) {
  const int lane = threadIdx.x & 31;
  const int ngrp = Ncols >> 6;  // 64-column groups per row strip
  long long wave = (long long)blockIdx.x * (blockDim.x >> 5) + (threadIdx.x >> 5);
  long long mt = wave / ngrp;
  int ng = (int)(wave % ngrp);
  if (mt * 16 >= M) return;

  const __bf16* Ab = A + (size_t)mt * 16 * K;
  v8f acc[4] = {};
  for (int k0 = 0; k0 < K; k0 += 32) {
    v16bf a = load_a_frag(Ab + k0, K, lane);
#pragma unroll
    for (int t = 0; t < 4; ++t) {
      v16bf b = load_b_frag(Bt + (size_t)(ng * 64 + t * 16) * K + k0, K, lane);
      acc[t] = __builtin_amdgcn_wmma_f32_16x16x32_bf16(
          false, a, false, b, (short)0, acc[t], false, false);
    }
  }
  // C/D layout: VGPR j -> row j (lanes 0-15), row j+8 (lanes 16-31); col = lane&15
  int r = (lane >> 4) * 8;
  int c = lane & 15;
#pragma unroll
  for (int t = 0; t < 4; ++t) {
    float* Cb = C + ((size_t)mt * 16 + r) * Ncols + ng * 64 + t * 16 + c;
#pragma unroll
    for (int j = 0; j < 8; ++j)
      Cb[(size_t)j * Ncols] = acc[t][j];
  }
}

// ---------------- attention coefficients: warp per (node, head) ----------------

__global__ void attn_coef_k(const float* __restrict__ h, const float* __restrict__ att_s,
                            const float* __restrict__ att_d, float* __restrict__ as_,
                            float* __restrict__ ad_, long long Nn, int H, int C) {
  int lane = threadIdx.x & 31;
  long long warp = (long long)blockIdx.x * (blockDim.x >> 5) + (threadIdx.x >> 5);
  if (warp >= Nn * H) return;
  long long n = warp / H;
  int hd = (int)(warp % H);
  const float* row = h + ((size_t)n * H + hd) * C;
  float s = 0.f, d = 0.f;
  for (int c = lane; c < C; c += 32) {
    float v = row[c];
    s += v * att_s[hd * C + c];
    d += v * att_d[hd * C + c];
  }
  for (int off = 16; off; off >>= 1) {
    s += __shfl_xor(s, off);
    d += __shfl_xor(d, off);
  }
  if (lane == 0) { as_[warp] = s; ad_[warp] = d; }
}

// ---------------- edge passes ----------------

__global__ void edge_score_k(const long long* __restrict__ ei, long long E, long long E2,
                             const float* __restrict__ as_, const float* __restrict__ ad_,
                             float* __restrict__ esc, unsigned* __restrict__ mmax, int H) {
  long long t = (long long)blockIdx.x * blockDim.x + threadIdx.x;
  if (t >= E2 * H) return;
  long long e = t / H;
  int hd = (int)(t % H);
  long long s, d;
  if (e < E) { s = ei[e]; d = ei[E + e]; } else { s = d = e - E; }
  float v = as_[s * H + hd] + ad_[d * H + hd];
  v = v > 0.f ? v : NEG_SLOPE * v;
  esc[t] = v;
  atomicMax(mmax + d * H + hd, ord_enc(v));
}

__global__ void edge_exp_k(const long long* __restrict__ ei, long long E, long long E2,
                           float* __restrict__ esc, const unsigned* __restrict__ mmax,
                           float* __restrict__ ssum, int H) {
  long long t = (long long)blockIdx.x * blockDim.x + threadIdx.x;
  if (t >= E2 * H) return;
  long long e = t / H;
  int hd = (int)(t % H);
  long long d = (e < E) ? ei[E + e] : e - E;
  float m = ord_dec(mmax[d * H + hd]);
  float w = __expf(esc[t] - m);
  esc[t] = w;
  atomicAdd(ssum + d * H + hd, w);
}

// warp per (edge, head): out[dst, hd, :] += h[src, hd, :] * alpha
__global__ void aggregate_k(const long long* __restrict__ ei, long long E, long long E2,
                            const float* __restrict__ h, const float* __restrict__ wgt,
                            const float* __restrict__ ssum, float* __restrict__ out,
                            int H, int C) {
  int lane = threadIdx.x & 31;
  long long warp = (long long)blockIdx.x * (blockDim.x >> 5) + (threadIdx.x >> 5);
  if (warp >= E2 * H) return;
  long long e = warp / H;
  int hd = (int)(warp % H);
  long long s, d;
  if (e < E) { s = ei[e]; d = ei[E + e]; } else { s = d = e - E; }
  float alpha = wgt[warp] / (ssum[d * H + hd] + 1e-16f);
  const float* hrow = h + ((size_t)s * H + hd) * C;
  float* orow = out + ((size_t)d * H + hd) * C;
  for (int c = lane; c < C; c += 32)
    atomicAdd(orow + c, hrow[c] * alpha);
}

// ---------------- epilogues ----------------

__global__ void elu_bias_cast_k(const float* __restrict__ in, const float* __restrict__ bias,
                                __bf16* __restrict__ outb, long long n, int C) {
  long long i = (long long)blockIdx.x * blockDim.x + threadIdx.x;
  if (i < n) {
    float v = in[i] + bias[i % C];
    v = v > 0.f ? v : expm1f(v);
    outb[i] = (__bf16)v;
  }
}

__global__ void add_bias_k(float* __restrict__ out, const float* __restrict__ bias,
                           long long n, int C) {
  long long i = (long long)blockIdx.x * blockDim.x + threadIdx.x;
  if (i < n) out[i] += bias[i % C];
}

// ---------------- launch ----------------

extern "C" void kernel_launch(void* const* d_in, const int* in_sizes, int n_in,
                              void* d_out, int out_size, void* d_ws, size_t ws_size,
                              hipStream_t stream) {
  const float*     x      = (const float*)d_in[0];
  const long long* ei     = (const long long*)d_in[1];
  const float*     W1     = (const float*)d_in[2];
  const float*     att_s1 = (const float*)d_in[3];
  const float*     att_d1 = (const float*)d_in[4];
  const float*     b1     = (const float*)d_in[5];
  const float*     W2     = (const float*)d_in[6];
  const float*     att_s2 = (const float*)d_in[7];
  const float*     att_d2 = (const float*)d_in[8];
  const float*     b2     = (const float*)d_in[9];
  (void)n_in; (void)out_size; (void)ws_size;

  const int F = 128, H1 = 4, C1 = 64, D1 = H1 * C1 /*256*/, C2 = 64;
  const long long N  = (long long)in_sizes[0] / F;
  const long long E  = (long long)in_sizes[1] / 2;
  const long long E2 = E + N;

  char* ws = (char*)d_ws;
  size_t off = 0;
  auto alloc = [&](size_t bytes) -> char* {
    char* p = ws + off;
    off = (off + bytes + 255) & ~(size_t)255;
    return p;
  };
  __bf16*   xb  = (__bf16*)  alloc((size_t)N * F * 2);
  __bf16*   w1t = (__bf16*)  alloc((size_t)D1 * F * 2);
  __bf16*   w2t = (__bf16*)  alloc((size_t)C2 * D1 * 2);
  float*    h1  = (float*)   alloc((size_t)N * D1 * 4);
  float*    as1 = (float*)   alloc((size_t)N * H1 * 4);
  float*    ad1 = (float*)   alloc((size_t)N * H1 * 4);
  unsigned* m1  = (unsigned*)alloc((size_t)N * H1 * 4);
  float*    s1  = (float*)   alloc((size_t)N * H1 * 4);
  float*    e1  = (float*)   alloc((size_t)E2 * H1 * 4);
  float*    o1  = (float*)   alloc((size_t)N * D1 * 4);
  __bf16*   h2b = (__bf16*)  alloc((size_t)N * D1 * 2);
  float*    t2  = (float*)   alloc((size_t)N * C2 * 4);
  float*    as2 = (float*)   alloc((size_t)N * 4);
  float*    ad2 = (float*)   alloc((size_t)N * 4);
  unsigned* m2  = (unsigned*)alloc((size_t)N * 4);
  float*    s2  = (float*)   alloc((size_t)N * 4);
  float*    e2  = (float*)   alloc((size_t)E2 * 4);

  float* outp = (float*)d_out;

  // per-call initialization (graph-capture safe)
  hipMemsetAsync(s1,   0, (size_t)N * H1 * 4, stream);
  hipMemsetAsync(s2,   0, (size_t)N * 4,      stream);
  hipMemsetAsync(o1,   0, (size_t)N * D1 * 4, stream);
  hipMemsetAsync(outp, 0, (size_t)N * C2 * 4, stream);
  { long long n = N * H1; fill_u32_k<<<(int)((n + 255) / 256), 256, 0, stream>>>(m1, NEGINF_ENC, n); }
  { long long n = N;      fill_u32_k<<<(int)((n + 255) / 256), 256, 0, stream>>>(m2, NEGINF_ENC, n); }

  // precision conversion
  { long long n = N * F; cast_bf16_k<<<(int)((n + 255) / 256), 256, 0, stream>>>(x, xb, n); }
  transpose_cast_k<<<(F * D1 + 255) / 256, 256, 0, stream>>>(W1, w1t, F, D1);
  transpose_cast_k<<<(D1 * C2 + 255) / 256, 256, 0, stream>>>(W2, w2t, D1, C2);

  // ---- layer 1 ----
  { long long waves = ((N + 15) / 16) * (D1 / 64);
    gemm_bf16_wmma_k<<<(int)((waves + 7) / 8), 256, 0, stream>>>(xb, w1t, h1, (int)N, D1, F); }
  { long long warps = N * H1;
    attn_coef_k<<<(int)((warps + 7) / 8), 256, 0, stream>>>(h1, att_s1, att_d1, as1, ad1, N, H1, C1); }
  { long long n = E2 * H1;
    edge_score_k<<<(int)((n + 255) / 256), 256, 0, stream>>>(ei, E, E2, as1, ad1, e1, m1, H1);
    edge_exp_k<<<(int)((n + 255) / 256), 256, 0, stream>>>(ei, E, E2, e1, m1, s1, H1);
    aggregate_k<<<(int)((n + 7) / 8), 256, 0, stream>>>(ei, E, E2, h1, e1, s1, o1, H1, C1); }
  { long long n = N * D1;
    elu_bias_cast_k<<<(int)((n + 255) / 256), 256, 0, stream>>>(o1, b1, h2b, n, D1); }

  // ---- layer 2 ----
  { long long waves = ((N + 15) / 16) * (C2 / 64);
    gemm_bf16_wmma_k<<<(int)((waves + 7) / 8), 256, 0, stream>>>(h2b, w2t, t2, (int)N, C2, D1); }
  { long long warps = N;
    attn_coef_k<<<(int)((warps + 7) / 8), 256, 0, stream>>>(t2, att_s2, att_d2, as2, ad2, N, 1, C2); }
  { long long n = E2;
    edge_score_k<<<(int)((n + 255) / 256), 256, 0, stream>>>(ei, E, E2, as2, ad2, e2, m2, 1);
    edge_exp_k<<<(int)((n + 255) / 256), 256, 0, stream>>>(ei, E, E2, e2, m2, s2, 1);
    aggregate_k<<<(int)((n + 7) / 8), 256, 0, stream>>>(ei, E, E2, t2, e2, s2, outp, 1, C2); }
  { long long n = N * C2;
    add_bias_k<<<(int)((n + 255) / 256), 256, 0, stream>>>(outp, b2, n, C2); }
}
